// S6_25907242729655
// MI455X (gfx1250) — matrix operand order repaired and checked by reading, not compiled, not run on hardware
//
#include <hip/hip_runtime.h>

// CDNA5 / gfx1250: wave32, WMMA 16x16x32 f16 -> f32
typedef __attribute__((ext_vector_type(16))) _Float16 v16h;
typedef __attribute__((ext_vector_type(8)))  _Float16 h8;
typedef __attribute__((ext_vector_type(8)))  float    v8f;
typedef __attribute__((ext_vector_type(4)))  int      v4i_t;

#define S6_D 1024            // model dim (d)
#define KCH 64               // k-chunk staged per round
#define NIT (S6_D / KCH)     // 16 staging rounds
#define PITCH 72             // halves per LDS row (144 B): r*36 mod 64 is a bank permutation
#define ABUF (64 * PITCH)    // halves per LDS A buffer

// Prefer the CDNA5 async global->LDS DMA path when the builtin exists.
#if defined(__has_builtin)
#if __has_builtin(__builtin_amdgcn_global_load_async_to_lds_b128)
#define S6_ASYNC 1
#endif
#endif
#ifndef S6_ASYNC
#define S6_ASYNC 0
#endif

#if S6_ASYNC
#define AS1 __attribute__((address_space(1)))
#define AS3 __attribute__((address_space(3)))
// builtin signature (from clang diagnostic): (AS1 v4i*, AS3 v4i*, imm int, imm int)
#define S6_GPTR(p) ((AS1 v4i_t*)(_Float16*)(p))
#define S6_LPTR(p) ((AS3 v4i_t*)(p))
#if __has_builtin(__builtin_amdgcn_s_wait_asynccnt)
#define S6_WAIT_ASYNC() __builtin_amdgcn_s_wait_asynccnt(0)
#else
#define S6_WAIT_ASYNC() asm volatile("s_wait_asynccnt 0x0" ::: "memory")
#endif
#endif

#define WMMA_F16(a, b, c) \
  __builtin_amdgcn_wmma_f32_16x16x32_f16(false, (a), false, (b), (short)0, (c), false, false)

// stable softplus using hardware v_exp_f32 / v_log_f32
__device__ __forceinline__ float sp_fast(float s) {
  return fmaxf(s, 0.0f) + __logf(1.0f + __expf(-fabsf(s)));
}

// ---------------------------------------------------------------------------
// Kernel 1: cb[row] = sum_n (x@W2^T + b2)_n * (x@W3^T + b3)_n
// ---------------------------------------------------------------------------
__global__ void __launch_bounds__(256)
s6_cb_kernel(const float* __restrict__ x,
             const float* __restrict__ W2, const float* __restrict__ b2,
             const float* __restrict__ W3, const float* __restrict__ b3,
             float* __restrict__ cb) {
  __shared__ float sB[16][17];
  __shared__ float sC[16][17];
  const int row = blockIdx.x;
  const int t   = threadIdx.x;
  const int n   = t & 15;
  const int kg  = t >> 4;
  const float* xr = x  + (size_t)row * S6_D + kg * 64;
  const float* w2 = W2 + (size_t)n   * S6_D + kg * 64;
  const float* w3 = W3 + (size_t)n   * S6_D + kg * 64;
  float pb = 0.f, pc = 0.f;
#pragma unroll
  for (int k = 0; k < 64; k += 4) {
    float4 xv = *(const float4*)(xr + k);
    float4 a  = *(const float4*)(w2 + k);
    float4 c  = *(const float4*)(w3 + k);
    pb += xv.x * a.x + xv.y * a.y + xv.z * a.z + xv.w * a.w;
    pc += xv.x * c.x + xv.y * c.y + xv.z * c.z + xv.w * c.w;
  }
  sB[n][kg] = pb;
  sC[n][kg] = pc;
  __syncthreads();
  if (t < 16) {
    float Bn = b2[t], Cn = b3[t];
#pragma unroll
    for (int j = 0; j < 16; ++j) { Bn += sB[t][j]; Cn += sC[t][j]; }
    sB[t][16] = Bn * Cn;
  }
  __syncthreads();
  if (t == 0) {
    float s = 0.f;
#pragma unroll
    for (int j = 0; j < 16; ++j) s += sB[j][16];
    cb[row] = s;
  }
}

// ---------------------------------------------------------------------------
// Kernel 2: f32 -> f16 conversion, 8 elements/thread, b128 stores
// ---------------------------------------------------------------------------
__global__ void __launch_bounds__(256)
s6_cvt_f16(const float* __restrict__ in, _Float16* __restrict__ out, int n8) {
  int i = blockIdx.x * 256 + threadIdx.x;
  if (i >= n8) return;
  const float4* p = (const float4*)in + (size_t)i * 2;
  float4 a = p[0], b = p[1];
  h8 o;
  o[0] = (_Float16)a.x; o[1] = (_Float16)a.y; o[2] = (_Float16)a.z; o[3] = (_Float16)a.w;
  o[4] = (_Float16)b.x; o[5] = (_Float16)b.y; o[6] = (_Float16)b.z; o[7] = (_Float16)b.w;
  ((h8*)out)[i] = o;
}

// ---------------------------------------------------------------------------
// A-fragment from LDS (contiguous layout: elems 0..15 = K+ahi+0..15)
// ---------------------------------------------------------------------------
__device__ __forceinline__ v16h frag_from_lds(const _Float16* p) {
  h8 a = *(const h8*)p;          // ds_load_b128
  h8 b = *(const h8*)(p + 8);    // ds_load_b128
  v16h v;
#pragma unroll
  for (int j = 0; j < 8; ++j) { v[j] = a[j]; v[j + 8] = b[j]; }
  return v;
}

__device__ __forceinline__ void s6_emit(float acc, float bias, int row, int col,
                                        const float* __restrict__ x,
                                        const float* __restrict__ cb,
                                        float* __restrict__ y) {
  float d = sp_fast(acc + bias);
  y[(size_t)row * S6_D + col] = x[(size_t)row * S6_D + col] * d * cb[row];
}

// ---------------------------------------------------------------------------
// Kernel 3 (main): y = x * softplus(x@W1^T + b1) * cb,  GEMM in f16 WMMA.
// Block: 8 waves, tile 64(M) x 256(N); wave tile 64(M) x 32(N), 8 accumulators.
// A chunk (64x64) double-buffered in LDS: next chunk's copy issued before the
// compute (async global->LDS DMA when available), one barrier per round.
// ---------------------------------------------------------------------------
__global__ void __launch_bounds__(256)
s6_wmma_f16(const _Float16* __restrict__ xh, const _Float16* __restrict__ w1h,
            const float* __restrict__ x, const float* __restrict__ b1,
            const float* __restrict__ cb, float* __restrict__ y) {
  __shared__ _Float16 sA[2 * ABUF];

  const int t    = threadIdx.x;        // 0..255
  const int lane = t & 31;
  const int wv   = t >> 5;             // 0..7
  const int r    = lane & 15;
  const int half = lane >> 4;          // 0 or 1
  const int ahi  = half * 16;
  const int mBase = blockIdx.x * 64;
  const int nBase = (blockIdx.y * 8 + wv) * 32;

  // staging: thread t copies 16 halves (32 B) of local row lr at col kc
  const int lr = t >> 2;
  const int kc = (t & 3) * 16;
  const _Float16* gA = xh + (size_t)(mBase + lr) * S6_D + kc;
  _Float16* stA = sA + lr * PITCH + kc;               // buffer-0 slot

  const _Float16* bp0 = w1h + (size_t)(nBase + r) * S6_D + half * 16;
  const _Float16* bp1 = bp0 + (size_t)16 * S6_D;

  v8f c00 = {}, c01 = {}, c10 = {}, c11 = {};
  v8f c20 = {}, c21 = {}, c30 = {}, c31 = {};

  // prologue: chunk 0 -> buffer 0
#if S6_ASYNC
  __builtin_amdgcn_global_load_async_to_lds_b128(S6_GPTR(gA),     S6_LPTR(stA),     0, 0);
  __builtin_amdgcn_global_load_async_to_lds_b128(S6_GPTR(gA + 8), S6_LPTR(stA + 8), 0, 0);
  S6_WAIT_ASYNC();
#else
  {
    h8 p0 = *(const h8*)(gA);
    h8 p1 = *(const h8*)(gA + 8);
    *(h8*)(stA) = p0;
    *(h8*)(stA + 8) = p1;
  }
#endif
  __syncthreads();

  for (int it = 0; it < NIT; ++it) {
    const _Float16* cur = sA + (it & 1) * ABUF;
    _Float16* nxt = stA + ((it + 1) & 1) * ABUF;
    const int kglob = it * KCH;
    const int knext = (kglob + KCH) & (S6_D - 1);     // wraps on last round (harmless)

    // issue next chunk's copy before the compute
#if S6_ASYNC
    __builtin_amdgcn_global_load_async_to_lds_b128(S6_GPTR(gA + knext),     S6_LPTR(nxt),     0, 0);
    __builtin_amdgcn_global_load_async_to_lds_b128(S6_GPTR(gA + knext + 8), S6_LPTR(nxt + 8), 0, 0);
#else
    h8 p0 = *(const h8*)(gA + knext);
    h8 p1 = *(const h8*)(gA + knext + 8);
#endif

#pragma unroll
    for (int kk = 0; kk < KCH; kk += 32) {
      v16h bf0 = *(const v16h*)(bp0 + kglob + kk);    // 2x global_load_b128 (L2-resident W1)
      v16h bf1 = *(const v16h*)(bp1 + kglob + kk);
      const _Float16* fb = cur + kk + ahi + r * PITCH;
      v16h f0 = frag_from_lds(fb + 0 * 16 * PITCH);
      v16h f1 = frag_from_lds(fb + 1 * 16 * PITCH);
      v16h f2 = frag_from_lds(fb + 2 * 16 * PITCH);
      v16h f3 = frag_from_lds(fb + 3 * 16 * PITCH);
      c00 = WMMA_F16(f0, bf0, c00);  c01 = WMMA_F16(f0, bf1, c01);
      c10 = WMMA_F16(f1, bf0, c10);  c11 = WMMA_F16(f1, bf1, c11);
      c20 = WMMA_F16(f2, bf0, c20);  c21 = WMMA_F16(f2, bf1, c21);
      c30 = WMMA_F16(f3, bf0, c30);  c31 = WMMA_F16(f3, bf1, c31);
    }

    // complete the copy, then sync (one barrier per round)
#if S6_ASYNC
    S6_WAIT_ASYNC();
#else
    *(h8*)(nxt) = p0;
    *(h8*)(nxt + 8) = p1;
#endif
    __syncthreads();
  }

  // Epilogue. C/D layout: VGPR j, lanes 0-15 -> M=j ; lanes 16-31 -> M=j+8.
  const int   col0  = nBase + r;
  const int   col1  = col0 + 16;
  const float bias0 = b1[col0];
  const float bias1 = b1[col1];
  const int   mOff  = mBase + half * 8;
#pragma unroll
  for (int jj = 0; jj < 8; ++jj) {
    const int row0 = mOff + jj;
    s6_emit(c00[jj], bias0, row0,      col0, x, cb, y);
    s6_emit(c01[jj], bias1, row0,      col1, x, cb, y);
    s6_emit(c10[jj], bias0, row0 + 16, col0, x, cb, y);
    s6_emit(c11[jj], bias1, row0 + 16, col1, x, cb, y);
    s6_emit(c20[jj], bias0, row0 + 32, col0, x, cb, y);
    s6_emit(c21[jj], bias1, row0 + 32, col1, x, cb, y);
    s6_emit(c30[jj], bias0, row0 + 48, col0, x, cb, y);
    s6_emit(c31[jj], bias1, row0 + 48, col1, x, cb, y);
  }
}

// ---------------------------------------------------------------------------
// Fallback (small workspace): on-the-fly f32->f16 convert, register-only GEMM.
// ---------------------------------------------------------------------------
__device__ __forceinline__ v16h load_a_frag_f32(const float* __restrict__ p) {
  v16h v;
#pragma unroll
  for (int j = 0; j < 8; ++j) v[j] = (_Float16)p[j];
#pragma unroll
  for (int j = 0; j < 8; ++j) v[j + 8] = (_Float16)p[8 + j];
  return v;
}
__device__ __forceinline__ v16h load_b_frag_f32(const float* __restrict__ p) {
  v16h v;
#pragma unroll
  for (int j = 0; j < 16; ++j) v[j] = (_Float16)p[j];
  return v;
}

__global__ void __launch_bounds__(128)
s6_wmma_fused_f32(const float* __restrict__ x, const float* __restrict__ W1,
                  const float* __restrict__ b1, const float* __restrict__ cb,
                  float* __restrict__ y) {
  const int lane = threadIdx.x & 31;
  const int wv   = threadIdx.x >> 5;
  const int r    = lane & 15;
  const int half = lane >> 4;
  const int nBase = (blockIdx.y * 4 + wv) * 16;
  const int mBase = blockIdx.x * 64;

  const float* bp = W1 + (size_t)(nBase + r) * S6_D + half * 16;
  const float* a0 = x  + (size_t)(mBase +  0 + r) * S6_D + half * 16;
  const float* a1 = a0 + (size_t)16 * S6_D;
  const float* a2 = a1 + (size_t)16 * S6_D;
  const float* a3 = a2 + (size_t)16 * S6_D;

  v8f c0 = {}, c1 = {}, c2 = {}, c3 = {};
  for (int k0 = 0; k0 < S6_D; k0 += 32) {
    v16h bf = load_b_frag_f32(bp + k0);
    v16h f0 = load_a_frag_f32(a0 + k0);
    v16h f1 = load_a_frag_f32(a1 + k0);
    v16h f2 = load_a_frag_f32(a2 + k0);
    v16h f3 = load_a_frag_f32(a3 + k0);
    c0 = WMMA_F16(f0, bf, c0);
    c1 = WMMA_F16(f1, bf, c1);
    c2 = WMMA_F16(f2, bf, c2);
    c3 = WMMA_F16(f3, bf, c3);
  }
  const int   col  = nBase + r;
  const float bias = b1[col];
  const int   mOff = mBase + half * 8;
#pragma unroll
  for (int jj = 0; jj < 8; ++jj) {
    const int row0 = mOff + jj;
    s6_emit(c0[jj], bias, row0,      col, x, cb, y);
    s6_emit(c1[jj], bias, row0 + 16, col, x, cb, y);
    s6_emit(c2[jj], bias, row0 + 32, col, x, cb, y);
    s6_emit(c3[jj], bias, row0 + 48, col, x, cb, y);
  }
}

// ---------------------------------------------------------------------------
// Inputs (setup_inputs order): x, W1, b1, W2, b2, W3, b3, A (A is dead).
// ---------------------------------------------------------------------------
extern "C" void kernel_launch(void* const* d_in, const int* in_sizes, int n_in,
                              void* d_out, int out_size, void* d_ws, size_t ws_size,
                              hipStream_t stream) {
  const float* x  = (const float*)d_in[0];
  const float* W1 = (const float*)d_in[1];
  const float* b1 = (const float*)d_in[2];
  const float* W2 = (const float*)d_in[3];
  const float* b2 = (const float*)d_in[4];
  const float* W3 = (const float*)d_in[5];
  const float* b3 = (const float*)d_in[6];
  float* y = (float*)d_out;

  const int rows = in_sizes[0] / S6_D;   // b*l = 8192
  char* ws = (char*)d_ws;

  float* cb = (float*)ws;
  size_t off = ((size_t)rows * 4 + 255) & ~(size_t)255;
  _Float16* xh  = (_Float16*)(ws + off);
  size_t off2 = off + (size_t)rows * S6_D * 2;
  _Float16* w1h = (_Float16*)(ws + off2);
  size_t need = off2 + (size_t)S6_D * S6_D * 2;

  s6_cb_kernel<<<rows, 256, 0, stream>>>(x, W2, b2, W3, b3, cb);

  if (need <= ws_size) {
    const int nx8 = rows * S6_D / 8;
    const int nw8 = S6_D * S6_D / 8;
    s6_cvt_f16<<<(nx8 + 255) / 256, 256, 0, stream>>>(x,  xh,  nx8);
    s6_cvt_f16<<<(nw8 + 255) / 256, 256, 0, stream>>>(W1, w1h, nw8);
    dim3 grid(rows / 64, S6_D / 256);    // (128, 4)
    s6_wmma_f16<<<grid, 256, 0, stream>>>(xh, w1h, x, b1, cb, y);
  } else {
    dim3 grid(rows / 64, S6_D / 64);     // (128, 16)
    s6_wmma_fused_f32<<<grid, 128, 0, stream>>>(x, W1, b1, cb, y);
  }
}
